// SigGNN_1580547975115
// MI455X (gfx1250) — matrix operand order repaired and checked by reading, not compiled, not run on hardware
//
#include <hip/hip_runtime.h>

#define NN 30490
#define EE 975680
#define NB ((NN + 15) / 16)

typedef __attribute__((ext_vector_type(16))) __bf16 bf16x16;
typedef __attribute__((ext_vector_type(8)))  float  v8f;

// ---------------------------------------------------------------- helpers
__device__ __forceinline__ int imin_(int a, int b){ return a < b ? a : b; }
__device__ __forceinline__ int iclamp_(int v, int lo, int hi){ return v < lo ? lo : (v > hi ? hi : v); }
__device__ __forceinline__ float fclip_(float v, float lo, float hi){ return fminf(fmaxf(v, lo), hi); }
__device__ __forceinline__ float nz_(float x){
  if (x != x) return 0.f;
  if (__builtin_isinf(x)) return x > 0.f ? 3.4028235e38f : -3.4028235e38f;
  return x;
}
__device__ __forceinline__ float gelu_(float x){ return 0.5f * x * (1.f + erff(x * 0.7071067811865475f)); }
__device__ __forceinline__ float softplus_(float x){ return x > 20.f ? x : log1pf(expf(x)); }
// monotone uint key for float atomic-max
__device__ __forceinline__ unsigned fkey_(float f){
  unsigned u = __float_as_uint(f);
  return (u & 0x80000000u) ? ~u : (u | 0x80000000u);
}
__device__ __forceinline__ float funkey_(unsigned k){
  unsigned u = (k & 0x80000000u) ? (k & 0x7FFFFFFFu) : ~k;
  return __uint_as_float(u);
}

union frag_u { uint4 q[2]; bf16x16 v; };

// One wave computes a 16x16 f32 tile: D += A(16 x 32*ksteps) * B(32*ksteps x 16).
// A: bf16 row-major in LDS, row stride lda (multiple of 8 elems, 16B-aligned base).
//    CDNA5 A-fragment per lane = two contiguous 8-elem runs -> 2x ds_load_b128.
// Bp: fragment-packed bf16 in workspace: per K-tile, per lane, 16 contiguous
//    elems (B[K = (e>>3)*16 + kh*8 + (e&7)][col = mn]) -> 2x global_load_b128, wave-coalesced.
__device__ __forceinline__ v8f wmma_bf16_tile(const __bf16* __restrict__ A, int lda,
                                              const __bf16* __restrict__ Bp,
                                              int ksteps, v8f acc)
{
  const int lid = threadIdx.x & 31;
  const int kh  = lid >> 4;     // K-half select
  const int mn  = lid & 15;     // row (A) / col (B)
  const uint4* Arow = reinterpret_cast<const uint4*>(A + (size_t)mn * lda);
  const uint4* Bl   = reinterpret_cast<const uint4*>(Bp + (size_t)lid * 16);
  for (int ks = 0; ks < ksteps; ++ks){
    frag_u au, bu;
    au.q[0] = Arow[ks * 4 + kh];          // K = ks*32 + kh*8 + 0..7
    au.q[1] = Arow[ks * 4 + 2 + kh];      // K = ks*32 + 16 + kh*8 + 0..7
    bu.q[0] = Bl[ks * 64 + 0];            // lane-packed 32B chunk
    bu.q[1] = Bl[ks * 64 + 1];
    if (ks + 1 < ksteps)
      __builtin_prefetch(Bp + (size_t)((ks + 1) * 32 + lid) * 16, 0, 3);
    acc = __builtin_amdgcn_wmma_f32_16x16x32_bf16(false, au.v, false, bu.v, (short)0, acc, false, false);
  }
  return acc;
}

// ---------------------------------------------------------------- weight pack: f32 row-major -> bf16 WMMA-fragment order
// dst[((ct*kTiles + kt)*32 + lane)*16 + e] = src[kt*32 + (e>>3)*16 + (lane>>4)*8 + (e&7)][ct*16 + (lane&15)]
__global__ __launch_bounds__(256)
void pack_bf16_kernel(const float* __restrict__ src, __bf16* __restrict__ dst,
                      int srows, int scols, int kTiles, int cTiles)
{
  const int i = blockIdx.x * blockDim.x + threadIdx.x;
  const int total = kTiles * cTiles * 512;
  if (i >= total) return;
  const int e  = i & 15;
  const int l  = (i >> 4) & 31;
  const int kt = (i >> 9) % kTiles;
  const int ct = i / (kTiles * 512);
  const int K = kt * 32 + (e >> 3) * 16 + (l >> 4) * 8 + (e & 7);
  const int C = ct * 16 + (l & 15);
  const float v = (K < srows && C < scols) ? src[K * scols + C] : 0.f;
  dst[i] = (__bf16)v;
}

// ---------------------------------------------------------------- node pipeline: LN + signatures + emb + fused matmul
__global__ __launch_bounds__(128)
void node_pipeline_kernel(const float* __restrict__ nf,
                          const float* __restrict__ in_g, const float* __restrict__ in_b,
                          const int* __restrict__ store_id, const int* __restrict__ dept_id,
                          const int* __restrict__ cat_id, const int* __restrict__ state_id,
                          const int* __restrict__ item_id,
                          const float* __restrict__ emb_store, const float* __restrict__ emb_dept,
                          const float* __restrict__ emb_cat, const float* __restrict__ emb_state,
                          const float* __restrict__ emb_item,
                          const __bf16* __restrict__ fusWp, const float* __restrict__ fus_b,
                          const float* __restrict__ fus_g, const float* __restrict__ fus_bb,
                          float* __restrict__ h_out)
{
  __shared__ float  xln[16 * 224];                    // 16 nodes x 28 t x 8 ch (layer-normed)
  __shared__ __attribute__((aligned(16))) __bf16 afrag[16 * 864];  // 816 sig + 40 emb + 8 pad
  __shared__ float  hbuf[16 * 64];
  __shared__ float  mu[16], rsd[16];
  const int tid  = threadIdx.x;
  const int base = blockIdx.x * 16;

  // Phase 1: per-(node,t) LayerNorm over 8 channels
  for (int r = tid; r < 16 * 28; r += 128){
    const int node = r / 28, t = r % 28;
    const int gn = imin_(base + node, NN - 1);
    const float* x = nf + (size_t)gn * 224 + t * 8;
    float m = 0.f;
#pragma unroll
    for (int c = 0; c < 8; ++c) m += x[c];
    m *= 0.125f;
    float var = 0.f;
#pragma unroll
    for (int c = 0; c < 8; ++c){ const float d = x[c] - m; var += d * d; }
    var *= 0.125f;
    const float rs = rsqrtf(var + 1e-5f);
#pragma unroll
    for (int c = 0; c < 8; ++c)
      xln[node * 224 + t * 8 + c] = nz_((x[c] - m) * rs * in_g[c] + in_b[c]);
  }
  __syncthreads();

  // Phase 2: depth-2 lead-lag signatures (8 threads/node, each owns rows 2j,2j+1 of s2)
  {
    const int node = tid >> 3, j = tid & 7;
    const float* xb = &xln[node * 224];
    for (int widx = 0; widx < 3; ++widx){
      const int w   = (widx == 0) ? 7 : ((widx == 1) ? 14 : 28);
      const int off = widx * 272;
      const int t0  = 28 - w;
      const int S   = 2 * w - 2;
      const int i0  = 2 * j, i1 = i0 + 1;
      float c0 = 0.f, c1 = 0.f;
      float s2a[16], s2b[16];
#pragma unroll
      for (int d = 0; d < 16; ++d){ s2a[d] = 0.f; s2b[d] = 0.f; }
      for (int t = 0; t < S; ++t){
        float dxv[16];
#pragma unroll
        for (int d = 0; d < 16; ++d){
          // lead (d<8): row (t+1)//2 ; lag (d>=8): row t//2
          const int r1 = (d < 8) ? ((t + 2) >> 1) : ((t + 1) >> 1);
          const int r0 = (d < 8) ? ((t + 1) >> 1) : (t >> 1);
          const int c  = d & 7;
          dxv[d] = xb[(t0 + r1) * 8 + c] - xb[(t0 + r0) * 8 + c];
        }
        const float ai = c0 + 0.5f * dxv[i0];
        const float bi = c1 + 0.5f * dxv[i1];
#pragma unroll
        for (int d = 0; d < 16; ++d){ s2a[d] += ai * dxv[d]; s2b[d] += bi * dxv[d]; }
        c0 += dxv[i0]; c1 += dxv[i1];
      }
      { // s1 = p[-1]-p[0] (lead & lag both reduce to x[w-1]-x[0])
        const int ca = i0 & 7, cb = i1 & 7;
        afrag[node * 864 + off + i0] = (__bf16)nz_(xb[(t0 + w - 1) * 8 + ca] - xb[t0 * 8 + ca]);
        afrag[node * 864 + off + i1] = (__bf16)nz_(xb[(t0 + w - 1) * 8 + cb] - xb[t0 * 8 + cb]);
      }
#pragma unroll
      for (int d = 0; d < 16; ++d){
        afrag[node * 864 + off + 16 + i0 * 16 + d] = (__bf16)nz_(s2a[d]);
        afrag[node * 864 + off + 16 + i1 * 16 + d] = (__bf16)nz_(s2b[d]);
      }
    }
  }

  // Phase 3: categorical embeddings (+ zero pad to 864)
  for (int k = tid; k < 16 * 48; k += 128){
    const int node = k / 48, d = k % 48;
    const int gn = imin_(base + node, NN - 1);
    float v = 0.f;
    if (d < 8)       v = emb_store[iclamp_(store_id[gn], 0, 9)    * 8  + d];
    else if (d < 16) v = emb_dept [iclamp_(dept_id[gn],  0, 6)    * 8  + (d - 8)];
    else if (d < 20) v = emb_cat  [iclamp_(cat_id[gn],   0, 2)    * 4  + (d - 16)];
    else if (d < 24) v = emb_state[iclamp_(state_id[gn], 0, 2)    * 4  + (d - 20)];
    else if (d < 40) v = emb_item [iclamp_(item_id[gn],  0, 3048) * 16 + (d - 24)];
    afrag[node * 864 + 816 + d] = (__bf16)v;
  }
  __syncthreads();

  // Phase 4: fused 16x864 @ 864x64 WMMA + GELU + LN epilogue
  const int wv = tid >> 5, lid = tid & 31, kh = lid >> 4, mn = lid & 15;
  v8f acc = {};
  acc = wmma_bf16_tile(afrag, 864, fusWp + (size_t)wv * 27 * 512, 27, acc);
#pragma unroll
  for (int r = 0; r < 8; ++r){
    const int m = r + 8 * kh, col = 16 * wv + mn;
    hbuf[m * 64 + col] = gelu_(acc[r] + fus_b[col]);
  }
  __syncthreads();
  if (tid < 16){
    float m = 0.f;
    for (int c = 0; c < 64; ++c) m += hbuf[tid * 64 + c];
    m *= (1.f / 64.f);
    float var = 0.f;
    for (int c = 0; c < 64; ++c){ const float d = hbuf[tid * 64 + c] - m; var += d * d; }
    var *= (1.f / 64.f);
    mu[tid] = m; rsd[tid] = rsqrtf(var + 1e-5f);
  }
  __syncthreads();
  for (int k = tid; k < 16 * 64; k += 128){
    const int node = k >> 6, c = k & 63;
    const int gn = base + node;
    if (gn < NN)
      h_out[(size_t)gn * 64 + c] = nz_((hbuf[k] - mu[node]) * rsd[node] * fus_g[c] + fus_bb[c]);
  }
}

// ---------------------------------------------------------------- GAT per-node projections: hs = h@Ws, hd = h@Wd, sS/sD dots
__global__ __launch_bounds__(128)
void gat_node_kernel(const float* __restrict__ h,
                     const __bf16* __restrict__ Wsp, const __bf16* __restrict__ Wdp,
                     const float* __restrict__ aS, const float* __restrict__ aD,
                     float* __restrict__ hs, float* __restrict__ sS, float* __restrict__ sD)
{
  __shared__ __attribute__((aligned(16))) __bf16 at[16 * 64];
  __shared__ float hsb[16 * 64], hdb[16 * 64];
  const int tid = threadIdx.x;
  const int base = blockIdx.x * 16;
  for (int k = tid; k < 16 * 64; k += 128){
    const int node = k >> 6, c = k & 63;
    const int gn = imin_(base + node, NN - 1);
    at[k] = (__bf16)h[(size_t)gn * 64 + c];
  }
  __syncthreads();
  const int wv = tid >> 5, lid = tid & 31, kh = lid >> 4, mn = lid & 15;
  v8f z = {};
  v8f a1 = wmma_bf16_tile(at, 64, Wsp + (size_t)wv * 2 * 512, 2, z);
  v8f a2 = wmma_bf16_tile(at, 64, Wdp + (size_t)wv * 2 * 512, 2, z);
#pragma unroll
  for (int r = 0; r < 8; ++r){
    const int m = r + 8 * kh, col = 16 * wv + mn;
    hsb[m * 64 + col] = a1[r];
    hdb[m * 64 + col] = a2[r];
  }
  __syncthreads();
  for (int k = tid; k < 16 * 64; k += 128){
    const int node = k >> 6, c = k & 63;
    const int gn = base + node;
    if (gn < NN) hs[(size_t)gn * 64 + c] = hsb[k];
  }
  for (int k = tid; k < 16 * 4; k += 128){
    const int node = k >> 2, hh = k & 3;
    const int gn = base + node;
    if (gn < NN){
      float s = 0.f, t = 0.f;
      for (int d = 0; d < 16; ++d){
        s += hsb[node * 64 + hh * 16 + d] * aS[hh * 16 + d];
        t += hdb[node * 64 + hh * 16 + d] * aD[hh * 16 + d];
      }
      sS[(size_t)gn * 4 + hh] = s;
      sD[(size_t)gn * 4 + hh] = t;
    }
  }
}

// ---------------------------------------------------------------- edge passes (segment softmax + aggregate)
__device__ __forceinline__ float edge_score(const float* sS, const float* sD, const float* tb,
                                            int s, int d, int ty, int hh)
{
  const float v = sS[(size_t)s * 4 + hh] + sD[(size_t)d * 4 + hh] + tb[ty * 4 + hh];
  return v > 0.f ? v : 0.2f * v;   // leaky_relu(0.2)
}

__global__ __launch_bounds__(256)
void edge_max_kernel(const int* __restrict__ ei, const int* __restrict__ et,
                     const float* __restrict__ sS, const float* __restrict__ sD,
                     const float* __restrict__ tb, unsigned* __restrict__ mkey)
{
  const int e = blockIdx.x * blockDim.x + threadIdx.x;
  if (e >= EE) return;
  const int s = ei[e], d = ei[EE + e], ty = et[e];
#pragma unroll
  for (int hh = 0; hh < 4; ++hh)
    atomicMax(&mkey[(size_t)d * 4 + hh], fkey_(edge_score(sS, sD, tb, s, d, ty, hh)));
}

__global__ __launch_bounds__(256)
void edge_sum_kernel(const int* __restrict__ ei, const int* __restrict__ et,
                     const float* __restrict__ sS, const float* __restrict__ sD,
                     const float* __restrict__ tb, const unsigned* __restrict__ mkey,
                     float* __restrict__ ssum)
{
  const int e = blockIdx.x * blockDim.x + threadIdx.x;
  if (e >= EE) return;
  const int s = ei[e], d = ei[EE + e], ty = et[e];
#pragma unroll
  for (int hh = 0; hh < 4; ++hh){
    const float v = edge_score(sS, sD, tb, s, d, ty, hh);
    atomicAdd(&ssum[(size_t)d * 4 + hh], expf(v - funkey_(mkey[(size_t)d * 4 + hh])));
  }
}

__global__ __launch_bounds__(256)
void edge_agg_kernel(const int* __restrict__ ei, const int* __restrict__ et,
                     const float* __restrict__ sS, const float* __restrict__ sD,
                     const float* __restrict__ tb, const unsigned* __restrict__ mkey,
                     const float* __restrict__ ssum, const float* __restrict__ hs,
                     float* __restrict__ agg)
{
  const int e = blockIdx.x * blockDim.x + threadIdx.x;
  if (e >= EE) return;
  const int s = ei[e], d = ei[EE + e], ty = et[e];
  float alpha[4];
#pragma unroll
  for (int hh = 0; hh < 4; ++hh){
    const float v = edge_score(sS, sD, tb, s, d, ty, hh);
    const float ex = expf(v - funkey_(mkey[(size_t)d * 4 + hh]));
    alpha[hh] = ex / (ssum[(size_t)d * 4 + hh] + 1e-9f);
  }
  const float4* hsv = reinterpret_cast<const float4*>(hs + (size_t)s * 64);
  float* dstp = agg + (size_t)d * 64;
#pragma unroll 4
  for (int q = 0; q < 16; ++q){
    const float4 vv = hsv[q];
    const float al = alpha[q >> 2];
    atomicAdd(dstp + q * 4 + 0, al * vv.x);
    atomicAdd(dstp + q * 4 + 1, al * vv.y);
    atomicAdd(dstp + q * 4 + 2, al * vv.z);
    atomicAdd(dstp + q * 4 + 3, al * vv.w);
  }
}

// ---------------------------------------------------------------- GAT output: h = LN(h + agg@Wo + bo)
__global__ __launch_bounds__(128)
void gat_out_kernel(float* __restrict__ h, const float* __restrict__ agg,
                    const __bf16* __restrict__ Wop, const float* __restrict__ bo,
                    const float* __restrict__ g, const float* __restrict__ b)
{
  __shared__ __attribute__((aligned(16))) __bf16 at[16 * 64];
  __shared__ float ob[16 * 64];
  __shared__ float mu[16], rsd[16];
  const int tid = threadIdx.x;
  const int base = blockIdx.x * 16;
  for (int k = tid; k < 16 * 64; k += 128){
    const int node = k >> 6, c = k & 63;
    const int gn = imin_(base + node, NN - 1);
    at[k] = (__bf16)agg[(size_t)gn * 64 + c];
  }
  __syncthreads();
  const int wv = tid >> 5, lid = tid & 31, kh = lid >> 4, mn = lid & 15;
  v8f z = {};
  v8f acc = wmma_bf16_tile(at, 64, Wop + (size_t)wv * 2 * 512, 2, z);
#pragma unroll
  for (int r = 0; r < 8; ++r){
    const int m = r + 8 * kh, col = 16 * wv + mn;
    ob[m * 64 + col] = acc[r];
  }
  __syncthreads();
  for (int k = tid; k < 16 * 64; k += 128){
    const int node = k >> 6, c = k & 63;
    const int gn = imin_(base + node, NN - 1);
    ob[k] = h[(size_t)gn * 64 + c] + ob[k] + bo[c];
  }
  __syncthreads();
  if (tid < 16){
    float m = 0.f;
    for (int c = 0; c < 64; ++c) m += ob[tid * 64 + c];
    m *= (1.f / 64.f);
    float var = 0.f;
    for (int c = 0; c < 64; ++c){ const float d = ob[tid * 64 + c] - m; var += d * d; }
    var *= (1.f / 64.f);
    mu[tid] = m; rsd[tid] = rsqrtf(var + 1e-5f);
  }
  __syncthreads();
  for (int k = tid; k < 16 * 64; k += 128){
    const int node = k >> 6, c = k & 63;
    const int gn = base + node;
    if (gn < NN)
      h[(size_t)gn * 64 + c] = nz_((ob[k] - mu[node]) * rsd[node] * g[c] + b[c]);
  }
}

// ---------------------------------------------------------------- predictor + reconciliation
__global__ __launch_bounds__(128)
void pred_kernel(const float* __restrict__ h,
                 const __bf16* __restrict__ W1p, const float* __restrict__ b1,
                 const float* __restrict__ pg, const float* __restrict__ pbln,
                 const __bf16* __restrict__ W2p, const float* __restrict__ b2,
                 const float* __restrict__ hscale, const float* __restrict__ grp_scale,
                 const int* __restrict__ dept_ids, const float* __restrict__ hmean,
                 float* __restrict__ out)
{
  __shared__ __attribute__((aligned(16))) __bf16 at[16 * 64];
  __shared__ float  zf[16 * 128];
  __shared__ __attribute__((aligned(16))) __bf16 zb[16 * 128];
  __shared__ float  pb[16 * 32];
  __shared__ float  mu[16], rsd[16];
  const int tid = threadIdx.x;
  const int base = blockIdx.x * 16;
  for (int k = tid; k < 16 * 64; k += 128){
    const int node = k >> 6, c = k & 63;
    const int gn = imin_(base + node, NN - 1);
    at[k] = (__bf16)h[(size_t)gn * 64 + c];
  }
  __syncthreads();
  const int wv = tid >> 5, lid = tid & 31, kh = lid >> 4, mn = lid & 15;
  // z = gelu(h @ W1 + b1): 8 column tiles across 4 waves
  for (int tile = wv; tile < 8; tile += 4){
    v8f z0 = {};
    v8f acc = wmma_bf16_tile(at, 64, W1p + (size_t)tile * 2 * 512, 2, z0);
#pragma unroll
    for (int r = 0; r < 8; ++r){
      const int m = r + 8 * kh, col = 16 * tile + mn;
      zf[m * 128 + col] = gelu_(acc[r] + b1[col]);
    }
  }
  __syncthreads();
  if (tid < 16){
    float m = 0.f;
    for (int c = 0; c < 128; ++c) m += zf[tid * 128 + c];
    m *= (1.f / 128.f);
    float var = 0.f;
    for (int c = 0; c < 128; ++c){ const float d = zf[tid * 128 + c] - m; var += d * d; }
    var *= (1.f / 128.f);
    mu[tid] = m; rsd[tid] = rsqrtf(var + 1e-5f);
  }
  __syncthreads();
  for (int k = tid; k < 16 * 128; k += 128){
    const int node = k >> 7, c = k & 127;
    zb[k] = (__bf16)nz_((zf[k] - mu[node]) * rsd[node] * pg[c] + pbln[c]);
  }
  __syncthreads();
  // pred = z @ W2 (+ b2 in epilogue): cols padded to 32, waves 0..1
  if (wv < 2){
    v8f z0 = {};
    v8f acc = wmma_bf16_tile(zb, 128, W2p + (size_t)wv * 4 * 512, 4, z0);
#pragma unroll
    for (int r = 0; r < 8; ++r){
      const int m = r + 8 * kh, col = 16 * wv + mn;
      pb[m * 32 + col] = acc[r];
    }
  }
  __syncthreads();
  if (tid < 16){
    const int gn = base + tid;
    if (gn < NN){
      const float gs = fclip_(softplus_(grp_scale[iclamp_(dept_ids[gn], 0, 6)]), 0.05f, 20.f);
      float pr[28];
      float mean = 0.f;
#pragma unroll
      for (int j = 0; j < 28; ++j){
        float x = pb[tid * 32 + j] + b2[j];
        x *= fclip_(softplus_(hscale[j]), 0.1f, 5.f);
        x = nz_(x);
        x *= gs;
        pr[j] = x;
        mean += x;
      }
      mean *= (1.f / 28.f);
      const float ratio = fclip_(hmean[gn] / (mean + 1e-6f), 0.05f, 20.f);
#pragma unroll
      for (int j = 0; j < 28; ++j)
        out[(size_t)gn * 28 + j] = fminf(fmaxf(pr[j] * ratio, 0.f), 1000.f);
    }
  }
}

// ---------------------------------------------------------------- host side
extern "C" void kernel_launch(void* const* d_in, const int* in_sizes, int n_in,
                              void* d_out, int out_size, void* d_ws, size_t ws_size,
                              hipStream_t stream)
{
  (void)in_sizes; (void)n_in; (void)out_size; (void)ws_size;
  // setup_inputs order, params flattened as sorted pytree leaves
  const float* nf        = (const float*)d_in[0];
  const int*   ei        = (const int*)  d_in[1];
  const int*   et        = (const int*)  d_in[2];
  const int*   store_id  = (const int*)  d_in[3];
  const int*   dept_id   = (const int*)  d_in[4];
  const int*   cat_id    = (const int*)  d_in[5];
  const int*   state_id  = (const int*)  d_in[6];
  const int*   item_id   = (const int*)  d_in[7];
  const int*   dept_ids  = (const int*)  d_in[8];
  const float* hmean     = (const float*)d_in[9];
  const float* emb_cat   = (const float*)d_in[10];
  const float* emb_dept  = (const float*)d_in[11];
  const float* emb_item  = (const float*)d_in[12];
  const float* emb_state = (const float*)d_in[13];
  const float* emb_store = (const float*)d_in[14];
  const float* fus_W     = (const float*)d_in[15];
  const float* fus_b     = (const float*)d_in[16];
  const float* fus_bb    = (const float*)d_in[17];
  const float* fus_g     = (const float*)d_in[18];
  // gat layers: base 19 + 9*l, keys sorted: Wd,Wo,Ws,aD,aS,b,bo,g,tb
  const float* grp_scale = (const float*)d_in[37];
  const float* hscale    = (const float*)d_in[38];
  const float* in_b      = (const float*)d_in[39];
  const float* in_g      = (const float*)d_in[40];
  const float* p_W1      = (const float*)d_in[41];
  const float* p_W2      = (const float*)d_in[42];
  const float* p_b1      = (const float*)d_in[43];
  const float* p_b2      = (const float*)d_in[44];
  const float* p_bln     = (const float*)d_in[45];
  const float* p_g       = (const float*)d_in[46];

  // workspace layout (floats); bf16 pool starts 16B-aligned (NN*208*4 bytes in)
  float* ws   = (float*)d_ws;
  float* h    = ws;
  float* hs   = h  + (size_t)NN * 64;
  float* sS   = hs + (size_t)NN * 64;
  float* sD   = sS + (size_t)NN * 4;
  unsigned* mkey = (unsigned*)(sD + (size_t)NN * 4);
  float* ssum = (float*)(mkey + (size_t)NN * 4);
  float* agg  = ssum + (size_t)NN * 4;
  __bf16* wb  = (__bf16*)(agg + (size_t)NN * 64);
  __bf16* fusWp = wb;                                 // 4 ct x 27 kt x 512 = 55296
  __bf16* W1p  = wb + 55296 + 2 * 12288;              // 8 ct x 2 kt x 512 = 8192
  __bf16* W2p  = W1p + 8192;                          // 2 ct x 4 kt x 512 = 4096

  // 1) pack weights into bf16 WMMA-fragment order
  pack_bf16_kernel<<<(864 * 64 + 255) / 256, 256, 0, stream>>>(fus_W, fusWp, 856, 64, 27, 4);
  for (int l = 0; l < 2; ++l){
    const int pbase = 19 + 9 * l;
    __bf16* Wsp = wb + 55296 + l * 12288;
    __bf16* Wdp = Wsp + 4096;
    __bf16* Wop = Wsp + 8192;
    pack_bf16_kernel<<<(64 * 64 + 255) / 256, 256, 0, stream>>>((const float*)d_in[pbase + 2], Wsp, 64, 64, 2, 4);
    pack_bf16_kernel<<<(64 * 64 + 255) / 256, 256, 0, stream>>>((const float*)d_in[pbase + 0], Wdp, 64, 64, 2, 4);
    pack_bf16_kernel<<<(64 * 64 + 255) / 256, 256, 0, stream>>>((const float*)d_in[pbase + 1], Wop, 64, 64, 2, 4);
  }
  pack_bf16_kernel<<<(64 * 128 + 255) / 256, 256, 0, stream>>>(p_W1, W1p, 64, 128, 2, 8);
  pack_bf16_kernel<<<(128 * 32 + 255) / 256, 256, 0, stream>>>(p_W2, W2p, 128, 28, 4, 2);

  // 2) node feature pipeline -> h
  node_pipeline_kernel<<<NB, 128, 0, stream>>>(nf, in_g, in_b,
      store_id, dept_id, cat_id, state_id, item_id,
      emb_store, emb_dept, emb_cat, emb_state, emb_item,
      fusWp, fus_b, fus_g, fus_bb, h);

  // 3) two GAT layers
  const int EB = (EE + 255) / 256;
  for (int l = 0; l < 2; ++l){
    const int pbase = 19 + 9 * l;
    const __bf16* Wsp = wb + 55296 + l * 12288;
    const __bf16* Wdp = Wsp + 4096;
    const __bf16* Wop = Wsp + 8192;
    const float* aD = (const float*)d_in[pbase + 3];
    const float* aS = (const float*)d_in[pbase + 4];
    const float* lb = (const float*)d_in[pbase + 5];
    const float* bo = (const float*)d_in[pbase + 6];
    const float* lg = (const float*)d_in[pbase + 7];
    const float* tb = (const float*)d_in[pbase + 8];

    // zero mkey + ssum + agg (contiguous: NN*4 + NN*4 + NN*64 words)
    hipMemsetAsync(mkey, 0, (size_t)NN * 72 * 4, stream);

    gat_node_kernel<<<NB, 128, 0, stream>>>(h, Wsp, Wdp, aS, aD, hs, sS, sD);
    edge_max_kernel<<<EB, 256, 0, stream>>>(ei, et, sS, sD, tb, mkey);
    edge_sum_kernel<<<EB, 256, 0, stream>>>(ei, et, sS, sD, tb, mkey, ssum);
    edge_agg_kernel<<<EB, 256, 0, stream>>>(ei, et, sS, sD, tb, mkey, ssum, hs, agg);
    gat_out_kernel<<<NB, 128, 0, stream>>>(h, agg, Wop, bo, lg, lb);
  }

  // 4) predictor + reconciliation -> out
  pred_kernel<<<NB, 128, 0, stream>>>(h, W1p, p_b1, p_g, p_bln, W2p, p_b2,
                                      hscale, grp_scale, dept_ids, hmean, (float*)d_out);
}
